// Transformer_13451837571790
// MI455X (gfx1250) — matrix operand order repaired and checked by reading, not compiled, not run on hardware
//
#include <hip/hip_runtime.h>
#include <hip/hip_bf16.h>

// ---------------------------------------------------------------------------
// Protein-RNA cross-attention transformer forward, MI455X (gfx1250, wave32).
// All GEMMs use v_wmma_f32_16x16x32_bf16 (bf16 in, f32 accumulate).
// Workspace use: ~130 MB (logits plane 67 MB dominates; everything L2-resident).
// ---------------------------------------------------------------------------

#define BB   8
#define SPP  512
#define SRR  101
#define DDIM 256
#define HH   8
#define DH   32      // DDIM / HH
#define DFFN 1024

typedef __attribute__((ext_vector_type(16))) __bf16 v16bf;
typedef __attribute__((ext_vector_type(8)))  float  v8f;

__device__ __forceinline__ int imin(int a, int b) { return a < b ? a : b; }

// ---------------------------------------------------------------------------
// Generic GEMM: C[M,N] = act(A[M,K] @ W[K,N] + bias[N]).
// One wave per 16x32 C slab (two 16x16 WMMA tiles sharing the A fragment).
// A row-major, W row-major. K % 32 == 0, N % 32 == 0; M arbitrary (OOB rows
// clamp-load valid data and are masked at store).
// ---------------------------------------------------------------------------
__global__ void wmma_gemm(const float* __restrict__ A, const float* __restrict__ W,
                          const float* __restrict__ bias, float* __restrict__ C,
                          int M, int N, int K, int relu)
{
  int wave = threadIdx.x >> 5;
  int lane = threadIdx.x & 31;
  int half = lane >> 4, mr = lane & 15;
  int tilesN2 = N >> 5;                         // 32-wide column slabs
  int tilesM = (M + 15) >> 4;
  int tile = blockIdx.x * 8 + wave;
  if (tile >= tilesM * tilesN2) return;         // wave-uniform exit
  int tm = tile / tilesN2, tn = tile % tilesN2;

  int arow = imin(tm * 16 + mr, M - 1);         // clamp: OOB rows masked at store
  const float* Ap  = A + (size_t)arow * K;
  const float* Wp0 = W + tn * 32 + mr;          // column slab, stride N
  const float* Wp1 = Wp0 + 16;

  v8f acc0 = {0.f,0.f,0.f,0.f,0.f,0.f,0.f,0.f};
  v8f acc1 = {0.f,0.f,0.f,0.f,0.f,0.f,0.f,0.f};
  for (int k0 = 0; k0 < K; k0 += 32) {
    v16bf af, b0, b1;
    // A 16x32 bf16 layout: half=0 -> elems 0..15: K 0..15; half=1 -> K 16..31
#pragma unroll
    for (int i = 0; i < 4; i++) {
      int kk = k0 + half * 16 + 2 * i;
      af[2*i]     = (__bf16)Ap[kk];
      af[2*i + 1] = (__bf16)Ap[kk + 1];
      af[8 + 2*i] = (__bf16)Ap[kk + 8];
      af[9 + 2*i] = (__bf16)Ap[kk + 9];
    }
    // B 32x16 bf16 layout: lanes 0-15 -> K 0..15 of col n; lanes 16-31 -> K 16..31
#pragma unroll
    for (int i = 0; i < 8; i++) {
      size_t r0 = (size_t)(k0 + half * 16 + 2 * i) * N;
      b0[2*i]     = (__bf16)Wp0[r0];
      b0[2*i + 1] = (__bf16)Wp0[r0 + N];
      b1[2*i]     = (__bf16)Wp1[r0];
      b1[2*i + 1] = (__bf16)Wp1[r0 + N];
    }
    acc0 = __builtin_amdgcn_wmma_f32_16x16x32_bf16(false, af, false, b0,
                                                   (short)0, acc0, false, false);
    acc1 = __builtin_amdgcn_wmma_f32_16x16x32_bf16(false, af, false, b1,
                                                   (short)0, acc1, false, false);
  }
  int col0 = tn * 32 + mr, col1 = col0 + 16;
  float bv0 = bias ? bias[col0] : 0.0f;
  float bv1 = bias ? bias[col1] : 0.0f;
#pragma unroll
  for (int r = 0; r < 8; r++) {
    int row = tm * 16 + r + 8 * half;           // C layout: VGPR r -> M=r+8*half
    if (row < M) {
      float v0 = acc0[r] + bv0;
      float v1 = acc1[r] + bv1;
      if (relu) { v0 = fmaxf(v0, 0.0f); v1 = fmaxf(v1, 0.0f); }
      C[(size_t)row * N + col0] = v0;
      C[(size_t)row * N + col1] = v1;
    }
  }
}

// ---------------------------------------------------------------------------
// logits[b,h,k,q] = (Kh . Qh)/sqrt(dh); optional stat-pot table + mask.
// One wave per 16x16 (k,q) tile; single WMMA (dh == 32). grid.y = b*H + h.
// OOB rows/cols clamp-load valid data; results masked at store.
// ---------------------------------------------------------------------------
__global__ void logits_kernel(const float* __restrict__ Kx, const float* __restrict__ Qx,
                              float* __restrict__ Lg,
                              const float* __restrict__ mask, int maskT,
                              const float* __restrict__ hb, const float* __restrict__ pi,
                              int tabT,
                              const float* __restrict__ w_at, const float* __restrict__ w_aug,
                              int Sk, int Sq)
{
  int wave = threadIdx.x >> 5, lane = threadIdx.x & 31;
  int half = lane >> 4, mr = lane & 15;
  int tk = (Sk + 15) >> 4, tq = (Sq + 15) >> 4;
  int tile = blockIdx.x * 8 + wave;
  if (tile >= tk * tq) return;                  // wave-uniform exit
  int tkm = tile / tq, tqn = tile % tq;
  int b = blockIdx.y / HH, h = blockIdx.y % HH;

  int krow = imin(tkm * 16 + mr, Sk - 1);
  int qcol = imin(tqn * 16 + mr, Sq - 1);
  const float* Kp = Kx + ((size_t)b * Sk + krow) * DDIM + h * DH;
  const float* Qp = Qx + ((size_t)b * Sq + qcol) * DDIM + h * DH;

  v16bf af, bq;
#pragma unroll
  for (int i = 0; i < 4; i++) {
    int kk = half * 16 + 2 * i;
    af[2*i]     = (__bf16)Kp[kk];
    af[2*i + 1] = (__bf16)Kp[kk + 1];
    af[8 + 2*i] = (__bf16)Kp[kk + 8];
    af[9 + 2*i] = (__bf16)Kp[kk + 9];
  }
#pragma unroll
  for (int i = 0; i < 8; i++) {
    int kk = half * 16 + 2 * i;
    bq[2*i]     = (__bf16)Qp[kk];
    bq[2*i + 1] = (__bf16)Qp[kk + 1];
  }
  v8f acc = {0.f,0.f,0.f,0.f,0.f,0.f,0.f,0.f};
  acc = __builtin_amdgcn_wmma_f32_16x16x32_bf16(false, af, false, bq,
                                                (short)0, acc, false, false);

  float wat = 1.0f, waug = 0.0f;
  if (hb) { wat = w_at[0]; waug = w_aug[0]; }
  const float* tab = (h < HH / 2) ? hb : pi;
  const float iscale = 0.17677669529663688f;    // 1/sqrt(32)

#pragma unroll
  for (int r = 0; r < 8; r++) {
    int k = tkm * 16 + r + 8 * half;
    int q = tqn * 16 + mr;
    if (k < Sk && q < Sq) {
      float v = acc[r] * iscale;
      if (hb) {
        size_t ti = (size_t)b * SPP * SRR +
                    (tabT ? ((size_t)q * SRR + k) : ((size_t)k * SRR + q));
        v = v * wat + waug * tab[ti];
      }
      if (mask) {
        size_t mi = (size_t)b * Sk * Sq +
                    (maskT ? ((size_t)q * Sk + k) : ((size_t)k * Sq + q));
        v += mask[mi] * (-1e9f);
      }
      Lg[(((size_t)b * HH + h) * Sk + k) * Sq + q] = v;
    }
  }
}

// ---------------------------------------------------------------------------
// 2D softmax over the whole Sk*Sq plane per (b,h), then * scale (device scalar)
// ---------------------------------------------------------------------------
__global__ void softmax2d_kernel(float* __restrict__ Lg, int n,
                                 const float* __restrict__ scale)
{
  float* p = Lg + (size_t)blockIdx.x * n;
  __shared__ float sh[256];
  int t = threadIdx.x;
  float m = -3.0e38f;
  for (int i = t; i < n; i += 256) m = fmaxf(m, p[i]);
  sh[t] = m; __syncthreads();
  for (int s = 128; s > 0; s >>= 1) { if (t < s) sh[t] = fmaxf(sh[t], sh[t + s]); __syncthreads(); }
  m = sh[0]; __syncthreads();
  float sum = 0.f;
  for (int i = t; i < n; i += 256) sum += __expf(p[i] - m);
  sh[t] = sum; __syncthreads();
  for (int s = 128; s > 0; s >>= 1) { if (t < s) sh[t] += sh[t + s]; __syncthreads(); }
  float sc = scale[0] / sh[0];
  for (int i = t; i < n; i += 256) p[i] = __expf(p[i] - m) * sc;
}

// ---------------------------------------------------------------------------
// ctx[b,q,h*32+d] = sum_k A[b,h,k,q] * V[b,k,h*32+d]   (WMMA over Sk)
// One wave per (16 q x 16 d) tile; grid.y = b*H + h, 2 d-tiles per head.
// A loads clamp (garbage killed by zeroed B or masked stores); B (V) values
// select-zeroed for OOB k since they feed *valid* outputs.
// ---------------------------------------------------------------------------
__global__ void ctx_kernel(const float* __restrict__ Lg, const float* __restrict__ V,
                           float* __restrict__ Cx, int Sk, int Sq)
{
  int wave = threadIdx.x >> 5, lane = threadIdx.x & 31;
  int half = lane >> 4, mr = lane & 15;
  int tq = (Sq + 15) >> 4;
  int tile = blockIdx.x * 8 + wave;
  if (tile >= tq * 2) return;                   // wave-uniform exit
  int tqm = tile >> 1, td = tile & 1;
  int b = blockIdx.y / HH, h = blockIdx.y % HH;
  const float* Lbase = Lg + ((size_t)b * HH + h) * (size_t)Sk * Sq;

  int qrow = imin(tqm * 16 + mr, Sq - 1);
  int ncol = td * 16 + mr;                      // d within head

  v8f acc = {0.f,0.f,0.f,0.f,0.f,0.f,0.f,0.f};
  for (int k0 = 0; k0 < Sk; k0 += 32) {
    v16bf af, bq;
#pragma unroll
    for (int i = 0; i < 4; i++) {
      int kk = k0 + half * 16 + 2 * i;
      af[2*i]     = (__bf16)Lbase[(size_t)imin(kk,     Sk - 1) * Sq + qrow];
      af[2*i + 1] = (__bf16)Lbase[(size_t)imin(kk + 1, Sk - 1) * Sq + qrow];
      af[8 + 2*i] = (__bf16)Lbase[(size_t)imin(kk + 8, Sk - 1) * Sq + qrow];
      af[9 + 2*i] = (__bf16)Lbase[(size_t)imin(kk + 9, Sk - 1) * Sq + qrow];
    }
#pragma unroll
    for (int i = 0; i < 8; i++) {
      int kk = k0 + half * 16 + 2 * i;
      float v0 = V[((size_t)b * Sk + imin(kk,     Sk - 1)) * DDIM + h * DH + ncol];
      float v1 = V[((size_t)b * Sk + imin(kk + 1, Sk - 1)) * DDIM + h * DH + ncol];
      bq[2*i]     = (__bf16)(kk     < Sk ? v0 : 0.f);
      bq[2*i + 1] = (__bf16)(kk + 1 < Sk ? v1 : 0.f);
    }
    acc = __builtin_amdgcn_wmma_f32_16x16x32_bf16(false, af, false, bq,
                                                  (short)0, acc, false, false);
  }
#pragma unroll
  for (int r = 0; r < 8; r++) {
    int q = tqm * 16 + r + 8 * half;
    if (q < Sq)
      Cx[((size_t)b * Sq + q) * DDIM + h * DH + td * 16 + mr] = acc[r];
  }
}

// ---------------------------------------------------------------------------
// Embedding + positional encoding: x = (emb[tok]*sqrt(D) + pe) * sqrt(D)/D
// ---------------------------------------------------------------------------
__global__ void embed_kernel(const int* __restrict__ tok, const float* __restrict__ emb,
                             float* __restrict__ x, int S)
{
  int idx = blockIdx.x * blockDim.x + threadIdx.x;
  int total = BB * S * DDIM;
  if (idx >= total) return;
  int d = idx % DDIM;
  int s = (idx / DDIM) % S;
  int b = idx / (DDIM * S);
  float di = (float)(d & ~1);
  float rate = (float)s * __powf(10000.0f, -di / (float)DDIM);
  float pe = (d & 1) ? __cosf(rate) : __sinf(rate);
  float e = emb[tok[b * S + s] * DDIM + d];
  x[idx] = (e * 16.0f + pe) * (16.0f / 256.0f);     // sqrt(256)=16
}

// ---------------------------------------------------------------------------
// out = LayerNorm(x + y) * g + be ; one block (256 thr) per row, D == 256
// ---------------------------------------------------------------------------
__global__ void add_ln_kernel(const float* __restrict__ x, const float* __restrict__ y,
                              const float* __restrict__ g, const float* __restrict__ be,
                              float* __restrict__ out)
{
  __shared__ float sh[DDIM];
  int row = blockIdx.x, d = threadIdx.x;
  size_t i = (size_t)row * DDIM + d;
  float v = x[i] + y[i];
  sh[d] = v; __syncthreads();
  for (int s = DDIM / 2; s > 0; s >>= 1) { if (d < s) sh[d] += sh[d + s]; __syncthreads(); }
  float mu = sh[0] * (1.0f / DDIM); __syncthreads();
  float c = v - mu;
  sh[d] = c * c; __syncthreads();
  for (int s = DDIM / 2; s > 0; s >>= 1) { if (d < s) sh[d] += sh[d + s]; __syncthreads(); }
  float var = sh[0] * (1.0f / DDIM);
  out[i] = c * rsqrtf(var + 1e-9f) * g[d] + be[d];
}

// scales[0] = count_nonzero(protok[0]); scales[1] = SR
__global__ void scales_kernel(const int* __restrict__ protok, float* __restrict__ scales)
{
  __shared__ int sh[256];
  int t = threadIdx.x, c = 0;
  for (int i = t; i < SPP; i += 256) c += (protok[i] != 0);
  sh[t] = c; __syncthreads();
  for (int s = 128; s > 0; s >>= 1) { if (t < s) sh[t] += sh[t + s]; __syncthreads(); }
  if (t == 0) { scales[0] = (float)sh[0]; scales[1] = (float)SRR; }
}

__global__ void pool_kernel(const float* __restrict__ xp, const float* __restrict__ xr,
                            float* __restrict__ pooled)
{
  int idx = blockIdx.x * blockDim.x + threadIdx.x;
  if (idx >= BB * DDIM) return;
  int b = idx / DDIM, d = idx % DDIM;
  float sp = 0.f, sr = 0.f;
  for (int s = 0; s < SPP; s++) sp += xp[((size_t)b * SPP + s) * DDIM + d];
  for (int s = 0; s < SRR; s++) sr += xr[((size_t)b * SRR + s) * DDIM + d];
  pooled[idx] = 0.5f * (sp / SPP + sr / SRR);
}

__global__ void final_kernel(const float* __restrict__ pooled, const float* __restrict__ Wf,
                             const float* __restrict__ bf, float* __restrict__ out)
{
  int b = threadIdx.x;
  if (b >= BB) return;
  float l0 = bf[0], l1 = bf[1];
  for (int d = 0; d < DDIM; d++) {
    float p = pooled[b * DDIM + d];
    l0 += p * Wf[d * 2 + 0];
    l1 += p * Wf[d * 2 + 1];
  }
  float m = fmaxf(l0, l1);
  float e0 = __expf(l0 - m), e1 = __expf(l1 - m);
  float inv = 1.0f / (e0 + e1);
  out[b * 2 + 0] = e0 * inv;
  out[b * 2 + 1] = e1 * inv;
}

// ---------------------------------------------------------------------------
// Host-side orchestration
// ---------------------------------------------------------------------------
namespace {

struct LayerW {
  const float *Wq,*bq,*Wk,*bk,*Wv,*bv,*Wo,*bo,*W1,*b1,*W2,*b2,*g1,*be1,*g2,*be2;
};

struct Scratch {
  float *q, *k, *v, *ctx, *att, *out1, *ffn, *hid, *lg, *pooled, *scales;
};

inline LayerW pick(void* const* din, int i) {
  LayerW p;
  p.Wq  = (const float*)din[8]  + (size_t)i * DDIM * DDIM;
  p.bq  = (const float*)din[9]  + (size_t)i * DDIM;
  p.Wk  = (const float*)din[10] + (size_t)i * DDIM * DDIM;
  p.bk  = (const float*)din[11] + (size_t)i * DDIM;
  p.Wv  = (const float*)din[12] + (size_t)i * DDIM * DDIM;
  p.bv  = (const float*)din[13] + (size_t)i * DDIM;
  p.Wo  = (const float*)din[14] + (size_t)i * DDIM * DDIM;
  p.bo  = (const float*)din[15] + (size_t)i * DDIM;
  p.W1  = (const float*)din[16] + (size_t)i * DDIM * DFFN;
  p.b1  = (const float*)din[17] + (size_t)i * DFFN;
  p.W2  = (const float*)din[18] + (size_t)i * DFFN * DDIM;
  p.b2  = (const float*)din[19] + (size_t)i * DDIM;
  p.g1  = (const float*)din[20] + (size_t)i * DDIM;
  p.be1 = (const float*)din[21] + (size_t)i * DDIM;
  p.g2  = (const float*)din[22] + (size_t)i * DDIM;
  p.be2 = (const float*)din[23] + (size_t)i * DDIM;
  return p;
}

inline void run_gemm(const float* A, const float* W, const float* bias, float* C,
                     int M, int N, int K, int relu, hipStream_t s) {
  int tiles = ((M + 15) / 16) * (N / 32);
  int blocks = (tiles + 7) / 8;
  wmma_gemm<<<dim3(blocks), dim3(256), 0, s>>>(A, W, bias, C, M, N, K, relu);
}

// Full attention + FFN layer. `out` may alias `qin` (qin is last read before
// the final write). For cross layers, hb/pi non-null enables table augment.
void attn_layer_run(const float* qin, const float* kvin, int Sq, int Sk,
                    const float* mask, int maskT,
                    const float* hb, const float* pi, int tabT,
                    const float* scale_ptr, const LayerW& w,
                    const float* w_at, const float* w_aug,
                    float* out, const Scratch& ws, hipStream_t s)
{
  int Mq = BB * Sq, Mk = BB * Sk;
  run_gemm(qin,  w.Wq, w.bq, ws.q, Mq, DDIM, DDIM, 0, s);
  run_gemm(kvin, w.Wk, w.bk, ws.k, Mk, DDIM, DDIM, 0, s);
  run_gemm(kvin, w.Wv, w.bv, ws.v, Mk, DDIM, DDIM, 0, s);

  int tk = (Sk + 15) / 16, tq = (Sq + 15) / 16;
  logits_kernel<<<dim3((tk * tq + 7) / 8, BB * HH), dim3(256), 0, s>>>(
      ws.k, ws.q, ws.lg, mask, maskT, hb, pi, tabT, w_at, w_aug, Sk, Sq);
  softmax2d_kernel<<<dim3(BB * HH), dim3(256), 0, s>>>(ws.lg, Sk * Sq, scale_ptr);
  ctx_kernel<<<dim3((tq * 2 + 7) / 8, BB * HH), dim3(256), 0, s>>>(
      ws.lg, ws.v, ws.ctx, Sk, Sq);

  run_gemm(ws.ctx, w.Wo, w.bo, ws.att, Mq, DDIM, DDIM, 0, s);
  add_ln_kernel<<<dim3(Mq), dim3(DDIM), 0, s>>>(qin, ws.att, w.g1, w.be1, ws.out1);
  run_gemm(ws.out1, w.W1, w.b1, ws.hid, Mq, DFFN, DDIM, 1, s);
  run_gemm(ws.hid,  w.W2, w.b2, ws.ffn, Mq, DDIM, DFFN, 0, s);
  add_ln_kernel<<<dim3(Mq), dim3(DDIM), 0, s>>>(ws.out1, ws.ffn, w.g2, w.be2, out);
}

} // namespace

extern "C" void kernel_launch(void* const* d_in, const int* in_sizes, int n_in,
                              void* d_out, int out_size, void* d_ws, size_t ws_size,
                              hipStream_t stream)
{
  (void)in_sizes; (void)n_in; (void)out_size; (void)ws_size;

  const int*   protok = (const int*)d_in[0];
  const int*   rnatok = (const int*)d_in[1];
  const float* self_pro_mask = (const float*)d_in[2];
  const float* cross_mask    = (const float*)d_in[3];
  const float* hb    = (const float*)d_in[4];
  const float* pi    = (const float*)d_in[5];
  const float* emb_p = (const float*)d_in[6];
  const float* emb_r = (const float*)d_in[7];
  const float* Wf    = (const float*)d_in[24];
  const float* bf    = (const float*)d_in[25];
  const float* w_aug = (const float*)d_in[26];
  const float* w_at  = (const float*)d_in[27];
  float* out = (float*)d_out;

  // Workspace layout (floats). All activation buffers padded to BB*SPP rows so
  // edge-tile OOB reads stay in-bounds; values are masked at store time.
  const size_t ACT = (size_t)BB * SPP * DDIM;         // 1,048,576 floats
  float* w0 = (float*)d_ws;
  float* xpA = w0;                 // persistent xp
  float* xrA = xpA + ACT;          // persistent xr
  float* xpB = xrA + ACT;          // cross double-buffer
  float* xrB = xpB + ACT;
  Scratch ws;
  ws.q    = xrB + ACT;
  ws.k    = ws.q   + ACT;
  ws.v    = ws.k   + ACT;
  ws.ctx  = ws.v   + ACT;
  ws.att  = ws.ctx + ACT;
  ws.out1 = ws.att + ACT;
  ws.ffn  = ws.out1 + ACT;
  ws.hid  = ws.ffn + ACT;                              // BB*SPP*DFFN
  ws.lg   = ws.hid + (size_t)BB * SPP * DFFN;          // BB*HH*SPP*SPP
  ws.pooled = ws.lg + (size_t)BB * HH * SPP * SPP;
  ws.scales = ws.pooled + (size_t)BB * DDIM;

  LayerW p0 = pick(d_in, 0), p1 = pick(d_in, 1), p2 = pick(d_in, 2), p3 = pick(d_in, 3);

  // scale scalars: scales[0]=count_p, scales[1]=SR
  scales_kernel<<<dim3(1), dim3(256), 0, stream>>>(protok, ws.scales);

  // embeddings
  {
    int tp = BB * SPP * DDIM, tr = BB * SRR * DDIM;
    embed_kernel<<<dim3((tp + 255) / 256), dim3(256), 0, stream>>>(protok, emb_p, xpA, SPP);
    embed_kernel<<<dim3((tr + 255) / 256), dim3(256), 0, stream>>>(rnatok, emb_r, xrA, SRR);
  }

  // 2x protein self-attention (shared weights p0), in-place
  for (int it = 0; it < 2; ++it)
    attn_layer_run(xpA, xpA, SPP, SPP, self_pro_mask, 0, nullptr, nullptr, 0,
                   ws.scales + 0, p0, w_at, w_aug, xpA, ws, stream);

  // 2x RNA self-attention (shared weights p1), in-place, no mask/table
  for (int it = 0; it < 2; ++it)
    attn_layer_run(xrA, xrA, SRR, SRR, nullptr, 0, nullptr, nullptr, 0,
                   ws.scales + 1, p1, w_at, w_aug, xrA, ws, stream);

  // 2x cross pair (shared weights p2 for xr_new, p3 for xp_new), double-buffered
  float* xpc = xpA; float* xrc = xrA;
  float* xpn = xpB; float* xrn = xrB;
  for (int it = 0; it < 2; ++it) {
    // xr_new = attn(q=xr, kv=xp): Sq=SR, Sk=SP, mask [B,SP,SR]=[B,Sk,Sq], table direct
    attn_layer_run(xrc, xpc, SRR, SPP, cross_mask, 0, hb, pi, 0,
                   ws.scales + 0, p2, w_at, w_aug, xrn, ws, stream);
    // xp_new = attn(q=xp, kv=xr): Sq=SP, Sk=SR, mask/table transposed
    attn_layer_run(xpc, xrc, SPP, SRR, cross_mask, 1, hb, pi, 1,
                   ws.scales + 1, p3, w_at, w_aug, xpn, ws, stream);
    float* t;
    t = xpc; xpc = xpn; xpn = t;
    t = xrc; xrc = xrn; xrn = t;
  }

  // pool + final softmax
  pool_kernel<<<dim3((BB * DDIM + 255) / 256), dim3(256), 0, stream>>>(xpc, xrc, ws.pooled);
  final_kernel<<<dim3(1), dim3(32), 0, stream>>>(ws.pooled, Wf, bf, out);
}
